// AttentionLayer_120259085047
// MI455X (gfx1250) — compile-verified, matrix operand order checked by reading, not run on hardware
//
#include <hip/hip_runtime.h>

// ---------------------------------------------------------------------------
// Problem constants (B=4, S=2048, D=1024, H=16, DH=64)
// ---------------------------------------------------------------------------
#define BATCH 4
#define SEQ   2048
#define DMODEL 1024
#define NHEAD 16
#define DHEAD 64
#define MROWS (BATCH * SEQ)        // 8192

typedef __attribute__((ext_vector_type(16))) __bf16        v16bf;
typedef __attribute__((ext_vector_type(8)))  float         v8f;
typedef __attribute__((ext_vector_type(4)))  unsigned int  v4u;
typedef __attribute__((ext_vector_type(8)))  int           v8i;
typedef __attribute__((ext_vector_type(4)))  int           v4i;

union Frag16 {
    v16bf bf;
    unsigned int u32[8];
    unsigned short u16[16];
};

__device__ __forceinline__ unsigned short f2bf(float f) {
    union { float f; unsigned u; } x;
    x.f = f;
    unsigned r = x.u + 0x7FFFu + ((x.u >> 16) & 1u);   // round-to-nearest-even
    return (unsigned short)(r >> 16);
}

// K-dim base offset of VGPR r for 16-bit A/B fragments (ISA 7.12.2 layout):
// VGPR 0..3 -> K = 2r (+8 for lanes 16..31), VGPR 4..7 -> K = 16+2(r-4) (+8)
__device__ __forceinline__ int kbase(int r, int grp) {
    return (r < 4 ? 2 * r : 16 + 2 * (r - 4)) + 8 * grp;
}

__device__ __forceinline__ v8f wmma_bf16(const Frag16& a, const Frag16& b, v8f c) {
    return __builtin_amdgcn_wmma_f32_16x16x32_bf16(
        /*neg_a=*/false, a.bf, /*neg_b=*/false, b.bf,
        /*c_mod=*/(short)0, c, /*reuse_a=*/false, /*reuse_b=*/false);
}

__device__ __forceinline__ v8f zero8() {
    v8f z = {0.f, 0.f, 0.f, 0.f, 0.f, 0.f, 0.f, 0.f};
    return z;
}

// ---------------------------------------------------------------------------
// TDM: issue a 2-D tensor_load_to_lds per the CDNA5 D# layout (ISA ch.8).
//   group0: [1:0]=count=1, [63:32]=lds byte addr, [120:64]=global byte addr,
//           [127:126]=type 2
//   group1: [17:16]=data_size(1 -> 2B), [20]=pad_enable, [24:22]=pad_interval,
//           [31:25]=pad_amount, [79:48]=tensor_dim0, [111:80]=tensor_dim1,
//           [127:112]=tile_dim0, [143:128]=tile_dim1, [207:160]=dim0_stride
// w0cfg carries the data_size/padding word; dims are all < 2^16 here.
// ---------------------------------------------------------------------------
__device__ __forceinline__ void tdm_load_2d(unsigned lds_off, const void* gptr,
                                            unsigned w0cfg,
                                            unsigned tdim0, unsigned tdim1,
                                            unsigned tile0, unsigned tile1,
                                            unsigned stride0) {
    unsigned long long ga = (unsigned long long)(size_t)gptr;
    v4u g0;
    g0.x = 1u;                                               // count = 1
    g0.y = lds_off;                                          // LDS byte address
    g0.z = (unsigned)ga;                                     // global addr lo
    g0.w = ((unsigned)(ga >> 32) & 0x1FFFFFFu) | 0x80000000u;// addr hi | type=2
    v8i g1;
    g1[0] = (int)w0cfg;              // mask=0, data_size, padding config
    g1[1] = (int)(tdim0 << 16);      // [63:48]  = tensor_dim0 lo16
    g1[2] = (int)(tdim1 << 16);      // [95:80]  = tensor_dim1 lo16
    g1[3] = (int)(tile0 << 16);      // [127:112]= tile_dim0
    g1[4] = (int)tile1;              // [143:128]= tile_dim1 (tile_dim2 = 0)
    g1[5] = (int)stride0;            // [191:160]= tensor_dim0_stride lo32
    g1[6] = 0;
    g1[7] = 0;                       // tensor_dim1_stride unused (2-D tile)
    v4i gz = {0, 0, 0, 0};
#if __clang_major__ >= 23
    v8i gz8 = {0, 0, 0, 0, 0, 0, 0, 0};
    __builtin_amdgcn_tensor_load_to_lds(g0, g1, gz, gz, gz8, 0);
#else
    __builtin_amdgcn_tensor_load_to_lds(g0, g1, gz, gz, 0);
#endif
}

// data_size=2B | pad_enable | pad after 32 dwords (64 bf16 row) by 1 dword
#define TDM_CFG_K  (0x10000u | 0x100000u | (4u << 22))
// data_size=2B | pad_enable | pad after 16 dwords (32 bf16 row) by 1 dword
#define TDM_CFG_V  (0x10000u | 0x100000u | (3u << 22))

// ---------------------------------------------------------------------------
// Kernel 1: QKV projection.  C[8192,1024] = X @ W + b, emitted as bf16 into
// per-head workspace layouts:  Q,K -> [B,H,S,DH] ; V -> [B,H,DH,S] (transposed)
// grid = (N/128, M/128, 3);  block = 256 threads (8 waves)
// ---------------------------------------------------------------------------
#define BM 128
#define BN 128
#define BK 32
#define LDT 34   // padded LDS K-stride (even -> u32-aligned pair reads)

__global__ __launch_bounds__(256)
void qkv_proj_kernel(const float* __restrict__ X,
                     const float* __restrict__ Wq, const float* __restrict__ bq,
                     const float* __restrict__ Wk, const float* __restrict__ bk,
                     const float* __restrict__ Wv, const float* __restrict__ bv,
                     unsigned short* __restrict__ Qws,
                     unsigned short* __restrict__ Kws,
                     unsigned short* __restrict__ Vws) {
    __shared__ __align__(16) unsigned short As[BM][LDT];   // [m][k] bf16
    __shared__ __align__(16) unsigned short Bs[BN][LDT];   // [n][k] bf16

    const int tid  = threadIdx.x;
    const int lane = tid & 31;
    const int idx  = lane & 15;
    const int grp  = lane >> 4;
    const int w    = tid >> 5;         // wave id 0..7
    const int wm   = w & 3;            // 4 waves along M
    const int wn   = w >> 2;           // 2 waves along N

    const int nblk = blockIdx.x * BN;
    const int mblk = blockIdx.y * BM;
    const int mode = blockIdx.z;       // 0=Q 1=K 2=V

    const float* W    = (mode == 0) ? Wq : (mode == 1) ? Wk : Wv;
    const float* bias = (mode == 0) ? bq : (mode == 1) ? bk : bv;
    unsigned short* dst = (mode == 0) ? Qws : (mode == 1) ? Kws : Vws;

    v8f acc[2][4];
#pragma unroll
    for (int mi = 0; mi < 2; ++mi)
#pragma unroll
        for (int ni = 0; ni < 4; ++ni) acc[mi][ni] = zero8();

    for (int kb = 0; kb < DMODEL; kb += BK) {
        // ---- stage X tile (128x32 f32 -> bf16) ----
#pragma unroll
        for (int it = 0; it < 4; ++it) {
            int lin = tid + it * 256;          // 0..1023
            int row = lin >> 3;                // 0..127
            int c4  = (lin & 7) * 4;           // 0..28
            float4 v = *(const float4*)(X + (size_t)(mblk + row) * DMODEL + kb + c4);
            unsigned p0 = (unsigned)f2bf(v.x) | ((unsigned)f2bf(v.y) << 16);
            unsigned p1 = (unsigned)f2bf(v.z) | ((unsigned)f2bf(v.w) << 16);
            *(unsigned*)&As[row][c4]     = p0;
            *(unsigned*)&As[row][c4 + 2] = p1;
        }
        // ---- stage W tile (32x128 f32 -> bf16, transposed into Bs[n][k]) ----
#pragma unroll
        for (int it = 0; it < 4; ++it) {
            int lin = tid + it * 256;
            int kr  = lin >> 5;                // 0..31
            int c4  = (lin & 31) * 4;          // 0..124
            float4 v = *(const float4*)(W + (size_t)(kb + kr) * DMODEL + nblk + c4);
            Bs[c4 + 0][kr] = f2bf(v.x);
            Bs[c4 + 1][kr] = f2bf(v.y);
            Bs[c4 + 2][kr] = f2bf(v.z);
            Bs[c4 + 3][kr] = f2bf(v.w);
        }
        __syncthreads();

        // ---- fragments + WMMA ----
        Frag16 af[2], bfr[4];
#pragma unroll
        for (int mi = 0; mi < 2; ++mi) {
            const unsigned short* p = &As[wm * 32 + mi * 16 + idx][0];
#pragma unroll
            for (int r = 0; r < 8; ++r)
                af[mi].u32[r] = *(const unsigned*)(p + kbase(r, grp));
        }
#pragma unroll
        for (int ni = 0; ni < 4; ++ni) {
            const unsigned short* p = &Bs[wn * 64 + ni * 16 + idx][0];
#pragma unroll
            for (int r = 0; r < 8; ++r)
                bfr[ni].u32[r] = *(const unsigned*)(p + kbase(r, grp));
        }
#pragma unroll
        for (int mi = 0; mi < 2; ++mi)
#pragma unroll
            for (int ni = 0; ni < 4; ++ni)
                acc[mi][ni] = wmma_bf16(af[mi], bfr[ni], acc[mi][ni]);
        __syncthreads();
    }

    // ---- epilogue: +bias, bf16, scatter into per-head layout ----
#pragma unroll
    for (int mi = 0; mi < 2; ++mi) {
#pragma unroll
        for (int ni = 0; ni < 4; ++ni) {
            int n = nblk + wn * 64 + ni * 16 + idx;
            float bv_ = bias[n];
            int h = n >> 6;          // head
            int d = n & 63;          // dim in head
#pragma unroll
            for (int r = 0; r < 8; ++r) {
                int m = mblk + wm * 32 + mi * 16 + r + 8 * grp;
                int b = m >> 11;     // /SEQ
                int s = m & 2047;
                unsigned short o = f2bf(acc[mi][ni][r] + bv_);
                if (mode == 2)       // V transposed [B,H,DH,S]
                    dst[((size_t)((b * NHEAD + h) * DHEAD + d)) * SEQ + s] = o;
                else                 // Q,K [B,H,S,DH]
                    dst[((size_t)((b * NHEAD + h) * SEQ + s)) * DHEAD + d] = o;
            }
        }
    }
}

// ---------------------------------------------------------------------------
// Kernel 2: flash attention with TDM double-buffered K/V staging.
// Each block = 8 waves sharing one (b,h); each wave owns 16 query rows.
// Wave 0 DMAs the next 32-key K tile (32x64, pad->ld 66) and V tile
// (64x32, pad->ld 34) into LDS via tensor_load_to_lds while all waves
// compute on the current tile; TENSORcnt + block barrier hand off.
// grid = (S/128, B*H); block = 256 threads
// ---------------------------------------------------------------------------
#define KLD 66
#define VLD 34

__global__ __launch_bounds__(256)
void flash_attn_kernel(const unsigned short* __restrict__ Qws,
                       const unsigned short* __restrict__ Kws,
                       const unsigned short* __restrict__ Vws,
                       const int* __restrict__ amask,
                       float* __restrict__ out) {
    __shared__ __align__(16) unsigned short Ks[2][32][KLD]; // [buf][kpos][dh]
    __shared__ __align__(16) unsigned short Vs[2][64][VLD]; // [buf][dh][kpos]
    __shared__ __align__(16) unsigned short Pl[8][16][LDT]; // per-wave P tile

    const int tid  = threadIdx.x;
    const int lane = tid & 31;
    const int idx  = lane & 15;
    const int grp  = lane >> 4;
    const int w    = tid >> 5;

    const int bh = blockIdx.y;
    const int b  = bh >> 4;
    const int h  = bh & 15;
    const int q0 = blockIdx.x * 128 + w * 16;

    const unsigned short* Qp = Qws + (size_t)bh * SEQ * DHEAD;
    const unsigned short* Kp = Kws + (size_t)bh * SEQ * DHEAD;
    const unsigned short* Vt = Vws + (size_t)bh * DHEAD * SEQ;  // [DH][S]
    const int* mp = amask + (size_t)b * SEQ;

    // Q fragments: 16 rows x DH=64 -> two K=32 A-fragments
    Frag16 qf[2];
#pragma unroll
    for (int f = 0; f < 2; ++f)
#pragma unroll
        for (int r = 0; r < 8; ++r)
            qf[f].u32[r] = *(const unsigned*)(Qp + (size_t)(q0 + idx) * DHEAD +
                                              f * 32 + kbase(r, grp));

    v8f o[4];
#pragma unroll
    for (int j = 0; j < 4; ++j) o[j] = zero8();
    float mrow[8], lrow[8];
#pragma unroll
    for (int r = 0; r < 8; ++r) { mrow[r] = -1e30f; lrow[r] = 0.f; }

    // prologue: DMA tile 0
    if (w == 0) {
        tdm_load_2d((unsigned)(size_t)&Ks[0][0][0], Kp, TDM_CFG_K,
                    /*tdim0=*/DHEAD, /*tdim1=*/32, /*tile0=*/DHEAD, /*tile1=*/32,
                    /*stride0=*/DHEAD);
        tdm_load_2d((unsigned)(size_t)&Vs[0][0][0], Vt, TDM_CFG_V,
                    /*tdim0=*/32, /*tdim1=*/DHEAD, /*tile0=*/32, /*tile1=*/DHEAD,
                    /*stride0=*/SEQ);
    }

    int cur = 0;
    for (int kb = 0; kb < SEQ; kb += 32, cur ^= 1) {
        __syncthreads();   // everyone done reading buf[cur^1] (prev iteration)
        if (w == 0) {
            if (kb + 32 < SEQ) {   // prefetch next tile into the other buffer
                tdm_load_2d((unsigned)(size_t)&Ks[cur ^ 1][0][0],
                            Kp + (size_t)(kb + 32) * DHEAD, TDM_CFG_K,
                            DHEAD, 32, DHEAD, 32, DHEAD);
                tdm_load_2d((unsigned)(size_t)&Vs[cur ^ 1][0][0],
                            Vt + (kb + 32), TDM_CFG_V,
                            32, DHEAD, 32, DHEAD, SEQ);
                __builtin_amdgcn_s_wait_tensorcnt(2);  // current tile resident
            } else {
                __builtin_amdgcn_s_wait_tensorcnt(0);
            }
        }
        __syncthreads();   // buf[cur] visible to all waves

        // ---- scores for 32 keys: two 16x16 fragments, K-dim = DH = 2x32 ----
        Frag16 kf;
        v8f s0 = zero8(), s1 = zero8();
#pragma unroll
        for (int f = 0; f < 2; ++f) {
#pragma unroll
            for (int r = 0; r < 8; ++r)
                kf.u32[r] = *(const unsigned*)&Ks[cur][idx][f * 32 + kbase(r, grp)];
            s0 = wmma_bf16(qf[f], kf, s0);
        }
#pragma unroll
        for (int f = 0; f < 2; ++f) {
#pragma unroll
            for (int r = 0; r < 8; ++r)
                kf.u32[r] = *(const unsigned*)&Ks[cur][16 + idx][f * 32 + kbase(r, grp)];
            s1 = wmma_bf16(qf[f], kf, s1);
        }

        const float madd0 = mp[kb + idx]      ? 0.f : -10000.f;
        const float madd1 = mp[kb + 16 + idx] ? 0.f : -10000.f;

        // ---- online softmax over this 32-key block ----
#pragma unroll
        for (int r = 0; r < 8; ++r) {
            float v0 = s0[r] * 0.125f + madd0;   // 1/sqrt(64)
            float v1 = s1[r] * 0.125f + madd1;
            float rm = fmaxf(v0, v1);
#pragma unroll
            for (int off = 8; off; off >>= 1)
                rm = fmaxf(rm, __shfl_xor(rm, off, 16));
            float nm   = fmaxf(mrow[r], rm);
            float corr = __expf(mrow[r] - nm);
            float p0   = __expf(v0 - nm);
            float p1   = __expf(v1 - nm);
            float ps   = p0 + p1;
#pragma unroll
            for (int off = 8; off; off >>= 1)
                ps += __shfl_xor(ps, off, 16);
            lrow[r] = lrow[r] * corr + ps;
            mrow[r] = nm;
            o[0][r] *= corr; o[1][r] *= corr; o[2][r] *= corr; o[3][r] *= corr;
            Pl[w][r + 8 * grp][idx]      = f2bf(p0);   // C-layout row -> LDS
            Pl[w][r + 8 * grp][16 + idx] = f2bf(p1);
        }
        __threadfence_block();   // order per-wave LDS write -> transposed read

        // ---- P as A-fragment (16q x 32k), V as B-fragments, accumulate ----
        Frag16 pf;
#pragma unroll
        for (int r = 0; r < 8; ++r)
            pf.u32[r] = *(const unsigned*)&Pl[w][idx][kbase(r, grp)];

#pragma unroll
        for (int j = 0; j < 4; ++j) {
            Frag16 vf;
#pragma unroll
            for (int r = 0; r < 8; ++r)
                vf.u32[r] = *(const unsigned*)&Vs[cur][j * 16 + idx][kbase(r, grp)];
            o[j] = wmma_bf16(pf, vf, o[j]);
        }
    }

    // ---- normalize + store fp32 output [B,S,D] ----
#pragma unroll
    for (int r = 0; r < 8; ++r) {
        float inv = 1.0f / lrow[r];
        int q = q0 + r + 8 * grp;
        float* orow = out + ((size_t)(b * SEQ + q)) * DMODEL + h * DHEAD;
#pragma unroll
        for (int j = 0; j < 4; ++j)
            orow[j * 16 + idx] = o[j][r] * inv;
    }
}

// ---------------------------------------------------------------------------
// Host-side launcher
// in:  0=inputs[B,S,D] f32, 1=attention_mask[B,S] i32,
//      2=Wq, 3=bq, 4=Wk, 5=bk, 6=Wv, 7=bv (f32)
// out: [B,S,D] f32 ; ws: 48 MB of bf16 Q/K/V
// ---------------------------------------------------------------------------
extern "C" void kernel_launch(void* const* d_in, const int* in_sizes, int n_in,
                              void* d_out, int out_size, void* d_ws, size_t ws_size,
                              hipStream_t stream) {
    const float* X    = (const float*)d_in[0];
    const int*   mask = (const int*)d_in[1];
    const float* Wq   = (const float*)d_in[2];
    const float* bq   = (const float*)d_in[3];
    const float* Wk   = (const float*)d_in[4];
    const float* bk   = (const float*)d_in[5];
    const float* Wv   = (const float*)d_in[6];
    const float* bv   = (const float*)d_in[7];
    float* out = (float*)d_out;

    unsigned short* Qws = (unsigned short*)d_ws;
    unsigned short* Kws = Qws + (size_t)MROWS * DMODEL;
    unsigned short* Vws = Kws + (size_t)MROWS * DMODEL;

    dim3 pgrid(DMODEL / BN, MROWS / BM, 3);     // (8, 64, 3)
    qkv_proj_kernel<<<pgrid, 256, 0, stream>>>(X, Wq, bq, Wk, bk, Wv, bv,
                                               Qws, Kws, Vws);

    dim3 agrid(SEQ / 128, BATCH * NHEAD);       // (16, 64)
    flash_attn_kernel<<<agrid, 256, 0, stream>>>(Qws, Kws, Vws, mask, out);
}
